// MoEBlock_42588895707847
// MI455X (gfx1250) — matrix operand order repaired and checked by reading, not compiled, not run on hardware
//
#include <hip/hip_runtime.h>
#include <hip/hip_bf16.h>
#include <math.h>

// ---------------- problem constants ----------------
#define NTOK 8192          // 4 * 2048 tokens
#define DDIM 1024
#define HDIM 4096
#define NEXP 8
#define MT   32            // tokens per FFN workgroup tile
#define HC   64            // H-chunk per fused iteration

// ---------------- WMMA vector types ----------------
typedef __attribute__((ext_vector_type(16))) __bf16 v16bf;
typedef __attribute__((ext_vector_type(8)))  __bf16 bf16x8;
typedef __attribute__((ext_vector_type(4)))  __bf16 bf16x4;
typedef __attribute__((ext_vector_type(8)))  float  v8f;

// ---------------- workspace layout (bytes) ----------------
#define OFF_CNT  0u
#define OFF_BASE 128u
#define OFF_IDX  256u
#define OFF_WTS  (256u + NEXP * NTOK * 4u)                         // 262400
#define OFF_W1T  (1u << 20)                                        // 1 MB
#define OFF_W2T  ((1u << 20) + (size_t)NEXP * HDIM * DDIM * 2u)    // 1MB + 64MB
#define OFF_XG   ((1u << 20) + (size_t)NEXP * HDIM * DDIM * 4u)    // 1MB + 128MB
// xg capacity: 16384 assignments + per-expert pad <= 16640 rows (~34 MB)

// zero out[] (as float4) and the per-expert counters
__global__ void zero_init_kernel(float4* __restrict__ out4, int* __restrict__ cnt) {
    size_t i = (size_t)blockIdx.x * blockDim.x + threadIdx.x;
    out4[i] = make_float4(0.f, 0.f, 0.f, 0.f);
    if (blockIdx.x == 0 && threadIdx.x < NEXP) cnt[threadIdx.x] = 0;
}

// generic 32x32 tiled transpose + f32->bf16 convert.
// z < NEXP: w1[e] (R=D rows, C=H cols) -> w1t[e] = [H][D]
// z >= NEXP: w2[e] (R=H rows, C=D cols) -> w2t[e] = [D][H]
__global__ void transpose_cvt_kernel(const float* __restrict__ w1,
                                     const float* __restrict__ w2,
                                     __bf16* __restrict__ w1t,
                                     __bf16* __restrict__ w2t) {
    int z = blockIdx.z;
    const float* src;
    __bf16* dst;
    int R, C;
    if (z < NEXP) {
        src = w1 + (size_t)z * DDIM * HDIM;
        dst = w1t + (size_t)z * HDIM * DDIM;
        R = DDIM; C = HDIM;
    } else {
        src = w2 + (size_t)(z - NEXP) * HDIM * DDIM;
        dst = w2t + (size_t)(z - NEXP) * DDIM * HDIM;
        R = HDIM; C = DDIM;
    }
    int r0 = blockIdx.y * 32, c0 = blockIdx.x * 32;
    if (r0 >= R || c0 >= C) return;
    __shared__ float tile[32][33];
    int tx = threadIdx.x & 31, ty = threadIdx.x >> 5;   // 32 x 8
#pragma unroll
    for (int i = 0; i < 32; i += 8)
        tile[ty + i][tx] = src[(size_t)(r0 + ty + i) * C + (c0 + tx)];
    __syncthreads();
#pragma unroll
    for (int i = 0; i < 32; i += 8)
        dst[(size_t)(c0 + ty + i) * R + (r0 + tx)] = (__bf16)tile[tx][ty + i];
}

// gating: per-token 8 logits, top-2, renormalized softmax weights,
// atomic-append into per-expert compacted lists.
__global__ void gate_kernel(const float* __restrict__ x,
                            const float* __restrict__ gw,   // [D][E]
                            const float* __restrict__ gb,   // [E]
                            int* __restrict__ cnt,
                            int* __restrict__ idx,          // [E][NTOK]
                            float* __restrict__ wts) {      // [E][NTOK]
    int t = blockIdx.x * blockDim.x + threadIdx.x;
    if (t >= NTOK) return;
    const float* xr = x + (size_t)t * DDIM;
    float acc[NEXP];
#pragma unroll
    for (int e = 0; e < NEXP; ++e) acc[e] = gb[e];
    for (int d = 0; d < DDIM; ++d) {
        float xv = xr[d];
        float4 g0 = *(const float4*)(gw + (size_t)d * NEXP);
        float4 g1 = *(const float4*)(gw + (size_t)d * NEXP + 4);
        acc[0] = fmaf(xv, g0.x, acc[0]); acc[1] = fmaf(xv, g0.y, acc[1]);
        acc[2] = fmaf(xv, g0.z, acc[2]); acc[3] = fmaf(xv, g0.w, acc[3]);
        acc[4] = fmaf(xv, g1.x, acc[4]); acc[5] = fmaf(xv, g1.y, acc[5]);
        acc[6] = fmaf(xv, g1.z, acc[6]); acc[7] = fmaf(xv, g1.w, acc[7]);
    }
    int i1 = 0;
#pragma unroll
    for (int e = 1; e < NEXP; ++e) if (acc[e] > acc[i1]) i1 = e;
    int i2 = (i1 == 0) ? 1 : 0;
#pragma unroll
    for (int e = 0; e < NEXP; ++e) if (e != i1 && acc[e] > acc[i2]) i2 = e;
    // softmax over all 8 then top-2 renorm == softmax over the top-2 logits
    float p1 = 1.f / (1.f + expf(acc[i2] - acc[i1]));
    float p2 = 1.f - p1;
    int pos1 = atomicAdd(&cnt[i1], 1);
    idx[i1 * NTOK + pos1] = t;  wts[i1 * NTOK + pos1] = p1;
    int pos2 = atomicAdd(&cnt[i2], 1);
    idx[i2 * NTOK + pos2] = t;  wts[i2 * NTOK + pos2] = p2;
}

// exclusive prefix sum of 32-row-padded expert counts -> xg row bases
__global__ void base_kernel(const int* __restrict__ cnt, int* __restrict__ base) {
    if (threadIdx.x == 0) {
        int run = 0;
        for (int e = 0; e < NEXP; ++e) {
            base[e] = run;
            run += (cnt[e] + MT - 1) & ~(MT - 1);
        }
    }
}

// gather routed tokens into compacted bf16 activation rows (tile tails zeroed)
__global__ void gather_kernel(const float* __restrict__ x,
                              const int* __restrict__ cnt,
                              const int* __restrict__ base,
                              const int* __restrict__ idx,
                              __bf16* __restrict__ xg) {
    int e = blockIdx.y;
    int slot = blockIdx.x;
    int ce = cnt[e];
    int pc = (ce + MT - 1) & ~(MT - 1);
    if (slot >= pc) return;
    int c0 = threadIdx.x * 4;
    __bf16* dst = xg + (size_t)(base[e] + slot) * DDIM + c0;
    bf16x4 o;
    if (slot < ce) {
        const float* srcr = x + (size_t)idx[e * NTOK + slot] * DDIM + c0;
        float4 v = *(const float4*)srcr;
        o[0] = (__bf16)v.x; o[1] = (__bf16)v.y;
        o[2] = (__bf16)v.z; o[3] = (__bf16)v.w;
    } else {
        o[0] = o[1] = o[2] = o[3] = (__bf16)0.f;
    }
    *(bf16x4*)dst = o;
}

// Fused expert FFN: per (expert, 32-token tile) workgroup of 8 waves.
// Loop over H in 64-chunks: phase A = gelu(xg @ w1 chunk) -> LDS (one 16x16
// WMMA tile per wave), phase B = y += h_chunk @ w2 chunk (wave owns
// 16 rows x 256 cols = 16 register-resident accumulator tiles).
__global__ void __launch_bounds__(256)
ffn_kernel(const __bf16* __restrict__ xg,    // compacted bf16 token rows
           const float* __restrict__ b1,     // [E][H]
           const float* __restrict__ b2,     // [E][D]
           const __bf16* __restrict__ w1t,   // [E][H][D] bf16
           const __bf16* __restrict__ w2t,   // [E][D][H] bf16
           const int* __restrict__ cnt,
           const int* __restrict__ base,
           const int* __restrict__ idx,
           const float* __restrict__ wts,
           float* __restrict__ out) {
    int e = blockIdx.y;
    int ce = cnt[e];
    int tbase = blockIdx.x * MT;
    if (tbase >= ce) return;

    __shared__ __bf16 h_s[MT][HC];   // 4 KB handoff tile
    __shared__ int   sidx[MT];
    __shared__ float swt[MT];

    int tid = threadIdx.x;
    if (tid < MT) {
        int slot = tbase + tid;
        bool v = slot < ce;
        sidx[tid] = v ? idx[e * NTOK + slot] : -1;
        swt[tid]  = v ? wts[e * NTOK + slot] : 0.f;
    }
    __syncthreads();

    int wv = tid >> 5, lane = tid & 31;
    int l16 = lane & 15, hlf = lane >> 4;
    int mi = wv & 1;              // 16-row sub-tile (both phases)
    int ni = wv >> 1;             // phase-A column sub-tile (0..3)
    int db = (wv >> 1) * 256;     // phase-B column base (0..768)

    const __bf16* w1e = w1t + (size_t)e * HDIM * DDIM;
    const __bf16* w2e = w2t + (size_t)e * DDIM * HDIM;
    const float*  b1e = b1 + (size_t)e * HDIM;
    const float*  b2e = b2 + (size_t)e * DDIM;

    // this lane's A row in the compacted bf16 activations (pad rows are zero)
    const __bf16* xrow = xg + (size_t)(base[e] + tbase + mi * 16 + l16) * DDIM;

    v8f yacc[16];
#pragma unroll
    for (int n2 = 0; n2 < 16; ++n2) yacc[n2] = {};

    const __bf16* hp = &h_s[mi * 16 + l16][0];   // phase-B A-fragment row

    for (int hc = 0; hc < HDIM; hc += HC) {
        // ---------------- phase A: h = gelu(x @ w1 + b1) ----------------
        int hb = hc + ni * 16;
        v8f hacc = {};
        const __bf16* bp = w1e + (size_t)(hb + l16) * DDIM + hlf * 16;
        for (int k0 = 0; k0 < DDIM; k0 += 32) {
            bf16x8 alo = *(const bf16x8*)(xrow + k0 + hlf * 16);
            bf16x8 ahi = *(const bf16x8*)(xrow + k0 + hlf * 16 + 8);
            v16bf af = __builtin_shufflevector(alo, ahi,
                0, 1, 2, 3, 4, 5, 6, 7, 8, 9, 10, 11, 12, 13, 14, 15);
            v16bf bfrag = *(const v16bf*)(bp + k0);
            hacc = __builtin_amdgcn_wmma_f32_16x16x32_bf16(
                false, af, false, bfrag, (short)0, hacc, false, false);
        }
        float bb = b1e[hb + l16];
#pragma unroll
        for (int j = 0; j < 8; ++j) {
            float v = hacc[j] + bb;
            float g = 0.5f * v * (1.0f + erff(v * 0.70710678118f));
            h_s[mi * 16 + j + 8 * hlf][ni * 16 + l16] = (__bf16)g;
        }
        __syncthreads();

        // ---------------- phase B: y += h @ w2 ----------------
#pragma unroll
        for (int ks = 0; ks < 2; ++ks) {
            int k0 = ks * 32;
            bf16x8 lo = *(const bf16x8*)(hp + k0 + hlf * 16);
            bf16x8 hi = *(const bf16x8*)(hp + k0 + hlf * 16 + 8);
            v16bf af = __builtin_shufflevector(lo, hi,
                0, 1, 2, 3, 4, 5, 6, 7, 8, 9, 10, 11, 12, 13, 14, 15);
            const __bf16* wp = w2e + hc + k0 + hlf * 16;
#pragma unroll
            for (int n2 = 0; n2 < 16; ++n2) {
                v16bf bfrag = *(const v16bf*)(wp + (size_t)(db + n2 * 16 + l16) * HDIM);
                yacc[n2] = __builtin_amdgcn_wmma_f32_16x16x32_bf16(
                    false, af, false, bfrag, (short)0, yacc[n2], false, false);
            }
        }
        __syncthreads();
    }

    // ---------------- epilogue: +b2, gate-scale, atomic combine ----------------
    int toks[8]; float wei[8];
#pragma unroll
    for (int j = 0; j < 8; ++j) {
        int m = mi * 16 + j + 8 * hlf;
        toks[j] = sidx[m];
        wei[j]  = swt[m];
    }
#pragma unroll
    for (int n2 = 0; n2 < 16; ++n2) {
        int col = db + n2 * 16 + l16;
        float bb2 = b2e[col];
#pragma unroll
        for (int j = 0; j < 8; ++j) {
            if (toks[j] >= 0)
                atomicAdd(out + (size_t)toks[j] * DDIM + col, wei[j] * (yacc[n2][j] + bb2));
        }
    }
}

extern "C" void kernel_launch(void* const* d_in, const int* in_sizes, int n_in,
                              void* d_out, int out_size, void* d_ws, size_t ws_size,
                              hipStream_t stream) {
    const float* x      = (const float*)d_in[0];
    const float* gate_w = (const float*)d_in[1];
    const float* gate_b = (const float*)d_in[2];
    const float* w1     = (const float*)d_in[3];
    const float* b1     = (const float*)d_in[4];
    const float* w2     = (const float*)d_in[5];
    const float* b2     = (const float*)d_in[6];
    float* out = (float*)d_out;

    char* wsb = (char*)d_ws;
    int*    cnt  = (int*)(wsb + OFF_CNT);
    int*    base = (int*)(wsb + OFF_BASE);
    int*    idx  = (int*)(wsb + OFF_IDX);
    float*  wts  = (float*)(wsb + OFF_WTS);
    __bf16* w1t  = (__bf16*)(wsb + OFF_W1T);
    __bf16* w2t  = (__bf16*)(wsb + OFF_W2T);
    __bf16* xg   = (__bf16*)(wsb + OFF_XG);

    // 1) zero output + routing counters
    zero_init_kernel<<<(NTOK * DDIM / 4) / 256, 256, 0, stream>>>((float4*)out, cnt);
    // 2) weight transpose + bf16 convert (w1 -> [E][H][D], w2 -> [E][D][H])
    transpose_cvt_kernel<<<dim3(HDIM / 32, HDIM / 32, 2 * NEXP), 256, 0, stream>>>(
        w1, w2, w1t, w2t);
    // 3) gating / routing
    gate_kernel<<<NTOK / 256, 256, 0, stream>>>(x, gate_w, gate_b, cnt, idx, wts);
    // 4) compacted row bases (prefix sum of padded counts)
    base_kernel<<<1, 32, 0, stream>>>(cnt, base);
    // 5) gather + convert routed tokens to bf16 rows (tile tails zeroed)
    gather_kernel<<<dim3(NTOK, NEXP), 256, 0, stream>>>(x, cnt, base, idx, xg);
    // 6) fused routed FFN (max 8192 tokens per expert -> 256 tile slots)
    ffn_kernel<<<dim3(NTOK / MT, NEXP), 256, 0, stream>>>(
        xg, b1, b2, w1t, w2t, cnt, base, idx, wts, out);
}
